// EncoderBlock_84799834293112
// MI455X (gfx1250) — matrix operand order repaired and checked by reading, not compiled, not run on hardware
//
#include <hip/hip_runtime.h>
#include <hip/hip_bf16.h>
#include <math.h>

// ---------------------------------------------------------------------------
// MI455X (gfx1250) encoder block: LN -> QKV -> flash-attn -> O+res -> LN ->
// FFN(ReLU)+res.  All GEMMs via v_wmma_f32_16x16x32_bf16 (wave32).
// Torch weights [N,K] row-major already match the WMMA B-matrix fragment
// order (fixed n, contiguous k) -> no transpose, just a bf16 cast.
// GEMM + attention use double-buffered LDS: global loads for tile i+1 are
// issued into registers before the WMMAs on tile i, stores go to the ping-
// pong buffer after compute, one barrier per tile.
// ---------------------------------------------------------------------------

typedef __bf16 bf16_t;
typedef __attribute__((ext_vector_type(16))) __bf16 v16bf;
typedef __attribute__((ext_vector_type(8)))  __bf16 v8bf;
typedef __attribute__((ext_vector_type(4)))  __bf16 v4bf;
typedef __attribute__((ext_vector_type(8)))  float  v8f;

__device__ __forceinline__ v16bf cat8(v8bf lo, v8bf hi) {
    return __builtin_shufflevector(lo, hi, 0,1,2,3,4,5,6,7,8,9,10,11,12,13,14,15);
}

__device__ __forceinline__ v8f wmma_bf16(v16bf a, v16bf b, v8f c) {
    // D = A(16x32 bf16) * B(32x16 bf16) + C(16x16 f32)
    return __builtin_amdgcn_wmma_f32_16x16x32_bf16(
        /*neg_a=*/false, a, /*neg_b=*/false, b,
        /*c_mod=*/(short)0, c, /*reuse_a=*/false, /*reuse_b=*/false);
}

// ---------------------------------------------------------------------------
// Elementwise f32 -> bf16 cast (weights stay in torch [N,K] layout).
// ---------------------------------------------------------------------------
__global__ __launch_bounds__(256) void cast_f32_to_bf16(
    const float* __restrict__ in, bf16_t* __restrict__ out)
{
    size_t i = ((size_t)blockIdx.x * 256 + threadIdx.x) * 4;
    float4 v = *(const float4*)(in + i);
    v4bf o = { (__bf16)v.x, (__bf16)v.y, (__bf16)v.z, (__bf16)v.w };
    *(v4bf*)(out + i) = o;
}

// ---------------------------------------------------------------------------
// LayerNorm (torch semantics: unbiased std, /(std+eps), scalar gamma/beta),
// output downconverted to bf16 for the following WMMA GEMM.
// ---------------------------------------------------------------------------
__global__ __launch_bounds__(256) void ln_kernel(
    const float* __restrict__ x, bf16_t* __restrict__ out,
    const float* __restrict__ g, const float* __restrict__ b, int Dd)
{
    __shared__ float rs[256], rs2[256];
    const int row = blockIdx.x, t = threadIdx.x;
    const float* xr = x + (size_t)row * Dd;
    float s = 0.f, s2 = 0.f;
    for (int i = t; i < Dd; i += 256) { float v = xr[i]; s += v; s2 += v * v; }
    rs[t] = s; rs2[t] = s2;
    __syncthreads();
    for (int off = 128; off > 0; off >>= 1) {
        if (t < off) { rs[t] += rs[t + off]; rs2[t] += rs2[t + off]; }
        __syncthreads();
    }
    float mean = rs[0] / (float)Dd;
    float var  = (rs2[0] - (float)Dd * mean * mean) / (float)(Dd - 1);
    var = fmaxf(var, 0.0f);
    float denom = sqrtf(var) + 1e-6f;
    float gg = g[0], bb = b[0];
    bf16_t* orow = out + (size_t)row * Dd;
    for (int i = t; i < Dd; i += 256)
        orow[i] = (bf16_t)(gg * (xr[i] - mean) / denom + bb);
}

// ---------------------------------------------------------------------------
// bf16 WMMA GEMM:  out[M,N] = A[M,K] @ W[N,K]^T (+bias)(+ReLU)(+res f32)
// BM=128, BN=128, BK=64; 8 waves in 4x2, each wave 32x64 (2x4 WMMA tiles).
// Double-buffered LDS, one barrier per K tile, 16 WMMAs per barrier.
// ---------------------------------------------------------------------------
__global__ __launch_bounds__(256) void gemm_bf16_kernel(
    const bf16_t* __restrict__ A,   // [M,K]
    const bf16_t* __restrict__ B,   // [N,K]  (weight)
    const float*  __restrict__ bias,
    const float*  __restrict__ residual,   // [M,N] f32 or null
    float*        __restrict__ outF,       // f32 out (or null)
    bf16_t*       __restrict__ outH,       // bf16 out (used when outF==null)
    int M, int N, int K, int doRelu)
{
    __shared__ alignas(16) bf16_t As[2][128][72];  // 64 K-halfs + 8 pad
    __shared__ alignas(16) bf16_t Bs[2][128][72];

    const int tid  = threadIdx.x;
    const int lane = tid & 31;
    const int w    = tid >> 5;
    const int wm   = w & 3, wn = w >> 2;
    const int mBase = blockIdx.y * 128;
    const int nBase = blockIdx.x * 128;
    const int m0 = wm * 32, n0 = wn * 64;

    const int laneRow = lane & 15;
    const int kOffA = (lane < 16) ? 0 : 8;    // A frag: lanes 16-31 hold K=8..15(+16)
    const int kOffB = (lane < 16) ? 0 : 16;   // unused: B frags also use kOffA (lanes 16-31 hold K=8..15,24..31)

    // per-thread staging geometry: 4 chunks of 8 halfs, 64B contiguous per row half
    const int cRow = (tid * 4) >> 3;          // row this thread stages
    const int cCol = ((tid * 4) & 7) * 8;     // starting half within the 64-K row
    const bf16_t* aSrc = A + (size_t)(mBase + cRow) * K + cCol;
    const bf16_t* bSrc = B + (size_t)(nBase + cRow) * K + cCol;

    v8f acc[2][4];
    #pragma unroll
    for (int i = 0; i < 2; ++i)
        #pragma unroll
        for (int j = 0; j < 4; ++j)
            acc[i][j] = (v8f){0.f,0.f,0.f,0.f,0.f,0.f,0.f,0.f};

    const int nk = K / 64;

    // ---- prologue: stage tile 0 ----
    v8bf ta[4], tb[4];
    #pragma unroll
    for (int j = 0; j < 4; ++j) {
        ta[j] = *(const v8bf*)(aSrc + j * 8);
        tb[j] = *(const v8bf*)(bSrc + j * 8);
    }
    #pragma unroll
    for (int j = 0; j < 4; ++j) {
        *(v8bf*)&As[0][cRow][cCol + j * 8] = ta[j];
        *(v8bf*)&Bs[0][cRow][cCol + j * 8] = tb[j];
    }
    __syncthreads();

    for (int it = 0; it < nk; ++it) {
        const int cur  = it & 1;
        const int nxt  = cur ^ 1;
        const int more = (it + 1) < nk;

        // ---- issue global loads for the NEXT tile (hidden under WMMAs) ----
        if (more) {
            const bf16_t* ap = aSrc + (size_t)(it + 1) * 64;
            const bf16_t* bp = bSrc + (size_t)(it + 1) * 64;
            #pragma unroll
            for (int j = 0; j < 4; ++j) {
                ta[j] = *(const v8bf*)(ap + j * 8);
                tb[j] = *(const v8bf*)(bp + j * 8);
            }
            if (it + 2 < nk) {  // warm L2 for tile i+2 (global_prefetch_b8)
                __builtin_prefetch(aSrc + (size_t)(it + 2) * 64, 0, 1);
                __builtin_prefetch(bSrc + (size_t)(it + 2) * 64, 0, 1);
            }
        }

        // ---- 16 WMMAs on the CURRENT tile ----
        #pragma unroll
        for (int ks = 0; ks < 64; ks += 32) {
            v16bf af[2], bfr[4];
            #pragma unroll
            for (int mt = 0; mt < 2; ++mt) {
                const bf16_t* p = &As[cur][m0 + mt * 16 + laneRow][ks + kOffA];
                af[mt] = cat8(*(const v8bf*)p, *(const v8bf*)(p + 16));
            }
            #pragma unroll
            for (int nt = 0; nt < 4; ++nt) {
                const bf16_t* p = &Bs[cur][n0 + nt * 16 + laneRow][ks + kOffA];
                bfr[nt] = cat8(*(const v8bf*)p, *(const v8bf*)(p + 16));
            }
            #pragma unroll
            for (int mt = 0; mt < 2; ++mt)
                #pragma unroll
                for (int nt = 0; nt < 4; ++nt)
                    acc[mt][nt] = wmma_bf16(af[mt], bfr[nt], acc[mt][nt]);
        }

        // ---- write next tile into the buffer nobody is reading ----
        if (more) {
            #pragma unroll
            for (int j = 0; j < 4; ++j) {
                *(v8bf*)&As[nxt][cRow][cCol + j * 8] = ta[j];
                *(v8bf*)&Bs[nxt][cRow][cCol + j * 8] = tb[j];
            }
        }
        __syncthreads();   // single barrier per K tile
    }

    // --- epilogue straight from C layout: row = r + (lane>=16?8:0), col = lane%16
    const int hi8 = (lane >= 16) ? 8 : 0;
    #pragma unroll
    for (int mt = 0; mt < 2; ++mt) {
        #pragma unroll
        for (int nt = 0; nt < 4; ++nt) {
            int gn = nBase + n0 + nt * 16 + laneRow;
            float bia = bias ? bias[gn] : 0.0f;
            #pragma unroll
            for (int r = 0; r < 8; ++r) {
                int gm = mBase + m0 + mt * 16 + hi8 + r;
                float v = acc[mt][nt][r] + bia;
                if (doRelu) v = fmaxf(v, 0.0f);
                size_t o = (size_t)gm * N + gn;
                if (residual) v += residual[o];
                if (outF) outF[o] = v;
                else      outH[o] = (bf16_t)v;
            }
        }
    }
}

// ---------------------------------------------------------------------------
// Flash attention, DK=64, key blocks of 32 (one WMMA K step).
// 8 waves/WG, 16 query rows per wave -> 128 rows per WG.
// scores: A = Q frag (registers), B = K^T (contiguous per-lane global loads)
// ctx:    A = P (per-wave LDS round-trip), B = V double-buffered in LDS.
// ---------------------------------------------------------------------------
__global__ __launch_bounds__(256) void attn_kernel(
    const bf16_t* __restrict__ qb, const bf16_t* __restrict__ kb,
    const bf16_t* __restrict__ vb, bf16_t* __restrict__ ob)
{
    constexpr int S = 2048, D = 1024, H = 16, DK = 64;
    __shared__ alignas(16) bf16_t Vt[2][64][40];    // V tile transposed [d][k]
    __shared__ alignas(16) bf16_t Pl[8][16][40];    // per-wave P tile 16x32

    const int tid = threadIdx.x, lane = tid & 31, w = tid >> 5;
    const int tilesPerBH = S / 128;
    const int bh = blockIdx.x / tilesPerBH;
    const int qt = blockIdx.x % tilesPerBH;
    const int b = bh / H, h = bh % H;
    const int q0 = qt * 128 + w * 16;

    const int laneRow = lane & 15;
    const int kOffA = (lane < 16) ? 0 : 8;
    const int kOffB = (lane < 16) ? 0 : 16;
    const int hi8   = (lane >= 16) ? 8 : 0;

    // V staging geometry: each thread copies 8 contiguous d-halfs of one key row
    const int kvl = tid >> 3;              // key row in block (0..31)
    const int d8  = (tid & 7) * 8;         // d offset
    const bf16_t* vSrc = vb + (size_t)(b * S + kvl) * D + h * DK + d8;

    // Q fragments (16 rows x 64 d) held in registers for the whole kernel
    v16bf qf[2];
    {
        const bf16_t* qp = qb + (size_t)(b * S + q0 + laneRow) * D + h * DK;
        #pragma unroll
        for (int dd = 0; dd < 2; ++dd) {
            const bf16_t* p = qp + dd * 32 + kOffA;
            qf[dd] = cat8(*(const v8bf*)p, *(const v8bf*)(p + 16));
        }
    }

    v8f acc[4];
    #pragma unroll
    for (int i = 0; i < 4; ++i) acc[i] = (v8f){0.f,0.f,0.f,0.f,0.f,0.f,0.f,0.f};
    float mrow[8], lsum[8];
    #pragma unroll
    for (int r = 0; r < 8; ++r) { mrow[r] = -3.0e38f; lsum[r] = 0.0f; }

    // ---- prologue: stage V block 0 ----
    v8bf vv = *(const v8bf*)vSrc;
    #pragma unroll
    for (int j = 0; j < 8; ++j) Vt[0][d8 + j][kvl] = vv[j];
    __syncthreads();

    for (int it = 0; it < S / 32; ++it) {
        const int cur  = it & 1;
        const int nxt  = cur ^ 1;
        const int more = (it + 1) < (S / 32);

        // ---- issue the next V block's global load early ----
        if (more) vv = *(const v8bf*)(vSrc + (size_t)(it + 1) * 32 * D);

        // ---- scores S = Q @ K^T  (B-matrix lanes read K rows contiguously) ----
        v8f s[2];
        s[0] = (v8f){0.f,0.f,0.f,0.f,0.f,0.f,0.f,0.f};
        s[1] = (v8f){0.f,0.f,0.f,0.f,0.f,0.f,0.f,0.f};
        #pragma unroll
        for (int dd = 0; dd < 2; ++dd) {
            #pragma unroll
            for (int nt = 0; nt < 2; ++nt) {
                const bf16_t* kp = kb
                    + (size_t)(b * S + it * 32 + nt * 16 + laneRow) * D
                    + h * DK + dd * 32 + kOffA;
                v16bf kf = cat8(*(const v8bf*)kp, *(const v8bf*)(kp + 16));
                s[nt] = wmma_bf16(qf[dd], kf, s[nt]);
            }
        }

        // ---- online softmax (rows live across 16 lanes of each half-wave) ----
        #pragma unroll
        for (int r = 0; r < 8; ++r) {
            float a0 = s[0][r] * 0.125f;  // 1/sqrt(64)
            float a1 = s[1][r] * 0.125f;
            float rm = fmaxf(a0, a1);
            rm = fmaxf(rm, __shfl_xor(rm, 1, 16));
            rm = fmaxf(rm, __shfl_xor(rm, 2, 16));
            rm = fmaxf(rm, __shfl_xor(rm, 4, 16));
            rm = fmaxf(rm, __shfl_xor(rm, 8, 16));
            float mn  = fmaxf(mrow[r], rm);
            float scl = __expf(mrow[r] - mn);
            float p0  = __expf(a0 - mn);
            float p1  = __expf(a1 - mn);
            float rs  = p0 + p1;
            rs += __shfl_xor(rs, 1, 16);
            rs += __shfl_xor(rs, 2, 16);
            rs += __shfl_xor(rs, 4, 16);
            rs += __shfl_xor(rs, 8, 16);
            lsum[r] = lsum[r] * scl + rs;
            mrow[r] = mn;
            #pragma unroll
            for (int nt = 0; nt < 4; ++nt) acc[nt][r] *= scl;
            // scatter P (C layout) to this wave's LDS tile
            Pl[w][hi8 + r][laneRow]      = (bf16_t)p0;
            Pl[w][hi8 + r][16 + laneRow] = (bf16_t)p1;
        }

        // ---- reload P as an A fragment (same-wave LDS ops complete in order) ----
        const bf16_t* pp = &Pl[w][laneRow][kOffA];
        v16bf pf = cat8(*(const v8bf*)pp, *(const v8bf*)(pp + 16));

        // ---- ctx += P @ V ----
        #pragma unroll
        for (int nt = 0; nt < 4; ++nt) {
            const bf16_t* vp2 = &Vt[cur][nt * 16 + laneRow][kOffA];
            v16bf vf = cat8(*(const v8bf*)vp2, *(const v8bf*)(vp2 + 16));
            acc[nt] = wmma_bf16(pf, vf, acc[nt]);
        }

        // ---- write next V block into the buffer nobody is reading ----
        if (more) {
            #pragma unroll
            for (int j = 0; j < 8; ++j) Vt[nxt][d8 + j][kvl] = vv[j];
        }
        __syncthreads();   // single barrier per key block
    }

    // ---- normalize and write ctx in [B,S,H,DK] (== [B*S, D]) bf16 ----
    #pragma unroll
    for (int nt = 0; nt < 4; ++nt) {
        #pragma unroll
        for (int r = 0; r < 8; ++r) {
            int gq = q0 + hi8 + r;
            size_t o = (size_t)(b * S + gq) * D + h * DK + nt * 16 + laneRow;
            ob[o] = (bf16_t)(acc[nt][r] / lsum[r]);
        }
    }
}

// ---------------------------------------------------------------------------
// Host-side orchestration
// ---------------------------------------------------------------------------
extern "C" void kernel_launch(void* const* d_in, const int* in_sizes, int n_in,
                              void* d_out, int out_size, void* d_ws, size_t ws_size,
                              hipStream_t stream)
{
    constexpr int B = 2, S = 2048, D = 1024, H = 16, DFF = 4096;
    constexpr int M = B * S;  // 4096 token rows

    const float* x   = (const float*)d_in[0];
    // d_in[1] = src_mask (no-op in reference)
    const float* wq  = (const float*)d_in[2];
    const float* bq  = (const float*)d_in[3];
    const float* wk  = (const float*)d_in[4];
    const float* bk  = (const float*)d_in[5];
    const float* wv  = (const float*)d_in[6];
    const float* bv  = (const float*)d_in[7];
    const float* wo  = (const float*)d_in[8];
    const float* bo  = (const float*)d_in[9];
    const float* w1  = (const float*)d_in[10];
    const float* b1  = (const float*)d_in[11];
    const float* w2  = (const float*)d_in[12];
    const float* b2  = (const float*)d_in[13];
    const float* g1  = (const float*)d_in[14];
    const float* be1 = (const float*)d_in[15];
    const float* g2  = (const float*)d_in[16];
    const float* be2 = (const float*)d_in[17];

    char* ws = (char*)d_ws;
    size_t off = 0;
    auto alloc = [&](size_t bytes) -> void* {
        void* p = ws + off;
        off += (bytes + 255) & ~(size_t)255;
        return p;
    };
    bf16_t* xn   = (bf16_t*)alloc((size_t)M * D * 2);      //  8 MB
    bf16_t* wqC  = (bf16_t*)alloc((size_t)D * D * 2);      //  2 MB (bf16, [N,K])
    bf16_t* wkC  = (bf16_t*)alloc((size_t)D * D * 2);
    bf16_t* wvC  = (bf16_t*)alloc((size_t)D * D * 2);
    bf16_t* woC  = (bf16_t*)alloc((size_t)D * D * 2);
    bf16_t* w1C  = (bf16_t*)alloc((size_t)DFF * D * 2);    //  8 MB
    bf16_t* w2C  = (bf16_t*)alloc((size_t)D * DFF * 2);    //  8 MB
    bf16_t* qbuf = (bf16_t*)alloc((size_t)M * D * 2);
    bf16_t* kbuf = (bf16_t*)alloc((size_t)M * D * 2);
    bf16_t* vbuf = (bf16_t*)alloc((size_t)M * D * 2);
    bf16_t* ctx  = (bf16_t*)alloc((size_t)M * D * 2);
    float*  x2   = (float*) alloc((size_t)M * D * 4);      // 16 MB
    bf16_t* xn2  = (bf16_t*)alloc((size_t)M * D * 2);
    bf16_t* hbuf = (bf16_t*)alloc((size_t)M * DFF * 2);    // 32 MB

    // weights: f32 -> bf16, keep torch [N,K] layout (matches WMMA B fragments)
    const int dd_blk  = (D * D) / 1024;        // 4 elems/thread, 256 threads
    const int dff_blk = (D * DFF) / 1024;
    cast_f32_to_bf16<<<dd_blk, 256, 0, stream>>>(wq, wqC);
    cast_f32_to_bf16<<<dd_blk, 256, 0, stream>>>(wk, wkC);
    cast_f32_to_bf16<<<dd_blk, 256, 0, stream>>>(wv, wvC);
    cast_f32_to_bf16<<<dd_blk, 256, 0, stream>>>(wo, woC);
    cast_f32_to_bf16<<<dff_blk, 256, 0, stream>>>(w1, w1C);
    cast_f32_to_bf16<<<dff_blk, 256, 0, stream>>>(w2, w2C);

    // --- residual 1: pre-norm attention ---
    ln_kernel<<<M, 256, 0, stream>>>(x, xn, g1, be1, D);

    gemm_bf16_kernel<<<dim3(D / 128, M / 128), 256, 0, stream>>>(
        xn, wqC, bq, nullptr, nullptr, qbuf, M, D, D, 0);
    gemm_bf16_kernel<<<dim3(D / 128, M / 128), 256, 0, stream>>>(
        xn, wkC, bk, nullptr, nullptr, kbuf, M, D, D, 0);
    gemm_bf16_kernel<<<dim3(D / 128, M / 128), 256, 0, stream>>>(
        xn, wvC, bv, nullptr, nullptr, vbuf, M, D, D, 0);

    attn_kernel<<<B * H * (S / 128), 256, 0, stream>>>(qbuf, kbuf, vbuf, ctx);

    gemm_bf16_kernel<<<dim3(D / 128, M / 128), 256, 0, stream>>>(
        ctx, woC, bo, x, x2, nullptr, M, D, D, 0);

    // --- residual 2: pre-norm FFN ---
    ln_kernel<<<M, 256, 0, stream>>>(x2, xn2, g2, be2, D);

    gemm_bf16_kernel<<<dim3(DFF / 128, M / 128), 256, 0, stream>>>(
        xn2, w1C, b1, nullptr, nullptr, hbuf, M, DFF, D, 1);

    gemm_bf16_kernel<<<dim3(D / 128, M / 128), 256, 0, stream>>>(
        hbuf, w2C, b2, x2, (float*)d_out, nullptr, M, D, DFF, 0);
}
